// MarkovSourceBP_73847667687631
// MI455X (gfx1250) — compile-verified, matrix operand order checked by reading, not run on hardware
//
#include <hip/hip_runtime.h>

typedef __attribute__((ext_vector_type(16))) _Float16 v16h;
typedef __attribute__((ext_vector_type(8)))  _Float16 v8h;
typedef __attribute__((ext_vector_type(4)))  _Float16 v4h;
typedef __attribute__((ext_vector_type(8)))  float    v8f;
typedef __attribute__((ext_vector_type(4)))  float    v4f;

#define HDIM 512
#define WDIM 512
#define MDIM 256
#define WM1  511          // W-1
#define ALPHA_F 0.9f
#define ELD 264           // epot LDS row pitch in f16 elems (528 B -> bank stride 4, conflict-free)
#define TLD 264           // temp LDS row pitch in f16 elems
#define RPB 128           // rows per block (8 waves x 16-row tiles)

// LDS layout (dynamic):
//   f16 epotS[256*ELD]   : epot row-major (row k, contiguous m) -> B fragments
//   f16 tempS[RPB*TLD]   : temp rows (contiguous m)             -> A fragments
//   f32 rowS [RPB]       : per-row sum of npot*Min (degenerate-belief check)
// total = 135168 + 67584 + 512 = 203264 B  (<= 320KB per WGP, 1 block/WGP)

extern "C" __global__ void __launch_bounds__(256)
bp_gemm(const float* __restrict__ Min, const float* __restrict__ npot,
        const float* __restrict__ epot, const float* __restrict__ Mprev,
        float* __restrict__ outNew, int dir)
{
    extern __shared__ char smem[];
    _Float16* epotS = (_Float16*)smem;
    _Float16* tempS = epotS + 256 * ELD;
    float*    rowS  = (float*)(tempS + RPB * TLD);

    const int tid = threadIdx.x;
    const long rowbase = (long)blockIdx.x * RPB;

    if (tid < RPB) rowS[tid] = 0.0f;
    __syncthreads();

    // ---- stage epot -> f16 LDS (row k contiguous in m) ----
    #pragma unroll 4
    for (int it = 0; it < 64; ++it) {
        int e = it * 256 + tid;           // vec4 group id
        int k = e >> 6;                   // 64 groups per 256-wide row
        int m = (e & 63) * 4;
        v4f e4 = *(const v4f*)(epot + (long)k * MDIM + m);
        v4h h4; h4[0]=(_Float16)e4[0]; h4[1]=(_Float16)e4[1];
                h4[2]=(_Float16)e4[2]; h4[3]=(_Float16)e4[3];
        *(v4h*)(epotS + (long)k * ELD + m) = h4;
    }

    // ---- stage temp = npot*Min*scale -> f16 LDS; accumulate S = sum(npot*Min) ----
    #pragma unroll 4
    for (int it = 0; it < 32; ++it) {
        int e  = it * 256 + tid;
        int rl = e >> 6;
        int m  = (e & 63) * 4;
        long gr = rowbase + rl;
        int h  = (int)(gr / WM1);
        int wc = (int)(gr % WM1);
        long pixBase, sclBase; bool useScale;
        if (dir == 0) {   // rightward: cols 0..W-2, scale = Mr[:, wc-1] for wc>0
            pixBase  = ((long)h * WDIM + wc) * MDIM;
            useScale = (wc != 0);
            sclBase  = ((long)h * WM1 + (wc - 1)) * MDIM;
        } else {          // leftward: cols 1..W-1, scale = Ml[:, wc+1] for wc<W-2
            pixBase  = ((long)h * WDIM + (wc + 1)) * MDIM;
            useScale = (wc != WM1 - 1);
            sclBase  = ((long)h * WM1 + (wc + 1)) * MDIM;
        }
        v4f n4 = *(const v4f*)(npot + pixBase + m);
        v4f i4 = *(const v4f*)(Min  + pixBase + m);
        v4f s4 = (v4f)1.0f;
        if (useScale) s4 = *(const v4f*)(Mprev + sclBase + m);
        v4f p4 = n4 * i4;
        atomicAdd(&rowS[rl], p4[0] + p4[1] + p4[2] + p4[3]);
        v4f t4 = p4 * s4;
        v4h h4; h4[0]=(_Float16)t4[0]; h4[1]=(_Float16)t4[1];
                h4[2]=(_Float16)t4[2]; h4[3]=(_Float16)t4[3];
        *(v4h*)(tempS + (long)rl * TLD + m) = h4;
    }
    __syncthreads();

    // ---- degenerate-belief fixup: where S==0, Min := 1/M  (never taken for these inputs) ----
    for (int it = 0; it < 32; ++it) {
        int e  = it * 256 + tid;
        int rl = e >> 6;
        if (rowS[rl] == 0.0f) {
            int m  = (e & 63) * 4;
            long gr = rowbase + rl;
            int h  = (int)(gr / WM1);
            int wc = (int)(gr % WM1);
            long pixBase, sclBase; bool useScale;
            if (dir == 0) { pixBase = ((long)h*WDIM + wc)*MDIM;     useScale = (wc != 0);       sclBase = ((long)h*WM1 + wc-1)*MDIM; }
            else          { pixBase = ((long)h*WDIM + wc+1)*MDIM;   useScale = (wc != WM1-1);   sclBase = ((long)h*WM1 + wc+1)*MDIM; }
            v4f n4 = *(const v4f*)(npot + pixBase + m);
            v4f s4 = (v4f)1.0f;
            if (useScale) s4 = *(const v4f*)(Mprev + sclBase + m);
            v4f t4 = n4 * s4 * (1.0f / (float)MDIM);
            v4h h4; h4[0]=(_Float16)t4[0]; h4[1]=(_Float16)t4[1];
                    h4[2]=(_Float16)t4[2]; h4[3]=(_Float16)t4[3];
            *(v4h*)(tempS + (long)rl * TLD + m) = h4;
        }
    }
    __syncthreads();

    // ---- GEMM: each wave owns a 16-row x 256-col output tile ----
    const int wave = tid >> 5;
    const int lane = tid & 31;
    const int half = lane >> 4;      // K-half selector per WMMA A/B layout
    const int nlo  = lane & 15;      // row (A) / column (B,D) within tile

    const _Float16* arow = tempS + (long)(wave * 16 + nlo) * TLD;

    v8f acc[16];
    #pragma unroll
    for (int t = 0; t < 16; ++t) acc[t] = (v8f)0.0f;

    #pragma unroll
    for (int kt = 0; kt < 8; ++kt) {
        const int mb = kt * 32;
        v8h alo = *(const v8h*)(arow + mb + 8 * half);        // K = mb + 8h .. +7
        v8h ahi = *(const v8h*)(arow + mb + 16 + 8 * half);   // K = mb+16+8h .. +7
        v16h a = __builtin_shufflevector(alo, ahi, 0,1,2,3,4,5,6,7,8,9,10,11,12,13,14,15);
        #pragma unroll
        for (int nt = 0; nt < 16; ++nt) {
            const _Float16* brow = epotS + (long)(nt * 16 + nlo) * ELD + mb;
            v8h blo = *(const v8h*)(brow + 8 * half);
            v8h bhi = *(const v8h*)(brow + 16 + 8 * half);
            v16h b = __builtin_shufflevector(blo, bhi, 0,1,2,3,4,5,6,7,8,9,10,11,12,13,14,15);
            acc[nt] = __builtin_amdgcn_wmma_f32_16x16x32_f16(
                false, a, false, b, (short)0, acc[nt], false, false);
        }
    }

    // ---- epilogue: row-normalize (fold in alpha), fuse EMA, store ----
    // C/D layout: lanes 0-15 hold rows v (VGPR v), lanes 16-31 hold rows v+8; N = nlo.
    float rinv[8];
    #pragma unroll
    for (int v = 0; v < 8; ++v) {
        float p = 0.0f;
        #pragma unroll
        for (int t = 0; t < 16; ++t) p += acc[t][v];
        p += __shfl_xor(p, 1, 32);
        p += __shfl_xor(p, 2, 32);
        p += __shfl_xor(p, 4, 32);
        p += __shfl_xor(p, 8, 32);   // masks < 16: reduction stays within each half
        rinv[v] = ALPHA_F / p;
    }
    #pragma unroll
    for (int v = 0; v < 8; ++v) {
        long gr   = rowbase + wave * 16 + v + 8 * half;
        long base = gr * MDIM + nlo;
        #pragma unroll
        for (int t = 0; t < 16; ++t) {
            long idx = base + t * 16;
            outNew[idx] = acc[t][v] * rinv[v] + (1.0f - ALPHA_F) * Mprev[idx];
        }
    }
}

// ---- beliefs: Mout = concat(1,Mr_new) * concat(Ml_new,1), row-normalized ----
extern "C" __global__ void __launch_bounds__(256)
bp_beliefs(const float* __restrict__ MrN, const float* __restrict__ MlN,
           float* __restrict__ Mout)
{
    const int wave = threadIdx.x >> 5;
    const int lane = threadIdx.x & 31;
    const long row = (long)blockIdx.x * 8 + wave;   // row = h*W + w
    const int h = (int)(row >> 9);
    const int w = (int)(row & (WDIM - 1));

    float vals[8];
    float sum = 0.0f;
    #pragma unroll
    for (int i = 0; i < 8; ++i) {
        int k = lane + 32 * i;
        float r = 1.0f, l = 1.0f;
        if (w > 0)        r = MrN[((long)h * WM1 + (w - 1)) * MDIM + k];
        if (w < WDIM - 1) l = MlN[((long)h * WM1 + w) * MDIM + k];
        vals[i] = r * l;
        sum += vals[i];
    }
    sum += __shfl_xor(sum, 1, 32);
    sum += __shfl_xor(sum, 2, 32);
    sum += __shfl_xor(sum, 4, 32);
    sum += __shfl_xor(sum, 8, 32);
    sum += __shfl_xor(sum, 16, 32);
    float inv = 1.0f / sum;
    #pragma unroll
    for (int i = 0; i < 8; ++i)
        Mout[row * MDIM + lane + 32 * i] = vals[i] * inv;
}

extern "C" void kernel_launch(void* const* d_in, const int* in_sizes, int n_in,
                              void* d_out, int out_size, void* d_ws, size_t ws_size,
                              hipStream_t stream) {
    const float* Min  = (const float*)d_in[0];
    const float* npot = (const float*)d_in[1];
    const float* epot = (const float*)d_in[2];
    const float* Mr   = (const float*)d_in[3];
    const float* Ml   = (const float*)d_in[4];

    float* Mout = (float*)d_out;                                   // H*W*M
    float* outR = Mout + (long)HDIM * WDIM * MDIM;                 // H*(W-1)*M
    float* outL = outR + (long)HDIM * WM1 * MDIM;                  // H*(W-1)*M

    const size_t shbytes = (size_t)(256 * ELD + RPB * TLD) * sizeof(_Float16)
                         + (size_t)RPB * sizeof(float);            // 203264 B
    static bool attrSet = false;
    (void)attrSet; // attribute call is idempotent & not a stream op; safe under capture
    hipFuncSetAttribute(reinterpret_cast<const void*>(&bp_gemm),
                        hipFuncAttributeMaxDynamicSharedMemorySize, (int)shbytes);

    const int nblocks = (HDIM * WM1) / RPB;                        // 2044
    bp_gemm<<<nblocks, 256, shbytes, stream>>>(Min, npot, epot, Mr, outR, 0);
    bp_gemm<<<nblocks, 256, shbytes, stream>>>(Min, npot, epot, Ml, outL, 1);
    bp_beliefs<<<(HDIM * WDIM) / 8, 256, 0, stream>>>(outR, outL, Mout);
}
